// GGMVAE2_48146583388765
// MI455X (gfx1250) — matrix operand order repaired and checked by reading, not compiled, not run on hardware
//
#include <hip/hip_runtime.h>

typedef __attribute__((ext_vector_type(16))) _Float16 v16h;
typedef __attribute__((ext_vector_type(8)))  _Float16 v8h;
typedef __attribute__((ext_vector_type(8)))  float    v8f;
typedef __attribute__((ext_vector_type(4)))  float    v4f;

#define DEV __device__ __forceinline__

// =====================================================================
// LDS tile helpers: As[64][40] (A, row-major), BsT[16][40] (B, transposed)
// 40*2B = 80B row stride keeps every v8h access 16B-aligned.
// =====================================================================
DEV v16h load_a_frag(const _Float16 (*As)[40], int wid, int lane)
{
    int m  = wid * 16 + (lane & 15);
    int kh = (lane >> 4) * 8;
    v8h lo = *(const v8h*)&As[m][kh];
    v8h hi = *(const v8h*)&As[m][16 + kh];
    return __builtin_shufflevector(lo, hi, 0,1,2,3,4,5,6,7,8,9,10,11,12,13,14,15);
}

DEV v16h load_b_frag(const _Float16 (*BsT)[40], int lane)
{
    int nn = lane & 15;
    int kb = (lane >> 4) * 8;
    v8h lo = *(const v8h*)&BsT[nn][kb];
    v8h hi = *(const v8h*)&BsT[nn][kb + 16];
    return __builtin_shufflevector(lo, hi, 0,1,2,3,4,5,6,7,8,9,10,11,12,13,14,15);
}

// cooperative B tile stage: 32x16 f16 chunk of Bp[K][ldb] -> BsT (transposed)
DEV void stage_b(const _Float16* __restrict__ Bp, int ldb, int k0, int nBase,
                 _Float16 (*BsT)[40], int tid)
{
    int e0 = tid * 4;
    int kr = e0 >> 4, nc0 = e0 & 15;
    const _Float16* bp = Bp + (size_t)(k0 + kr) * ldb + nBase + nc0;
    #pragma unroll
    for (int i = 0; i < 4; ++i) BsT[nc0 + i][kr] = bp[i];
}

// =====================================================================
// conv1 only: direct conv, k=4, s=2, p=1 (fp32, thread-per-output)
// =====================================================================
__global__ void conv_s2_kernel(const float* __restrict__ in, const float* __restrict__ w,
                               const float* __restrict__ bias, float* __restrict__ out,
                               int B, int CIN, int COUT, int HIN, int HOUT)
{
    int idx = blockIdx.x * blockDim.x + threadIdx.x;
    int total = B * COUT * HOUT * HOUT;
    if (idx >= total) return;
    int ox = idx % HOUT; int t = idx / HOUT;
    int oy = t % HOUT;   t /= HOUT;
    int o  = t % COUT;   int b = t / COUT;
    float s = bias[o];
    for (int c = 0; c < CIN; ++c) {
        const float* ip = in + (size_t)(b * CIN + c) * HIN * HIN;
        const float* wp = w  + (size_t)(o * CIN + c) * 16;
        #pragma unroll
        for (int ky = 0; ky < 4; ++ky) {
            int iy = oy * 2 - 1 + ky;
            if ((unsigned)iy >= (unsigned)HIN) continue;
            #pragma unroll
            for (int kx = 0; kx < 4; ++kx) {
                int ix = ox * 2 - 1 + kx;
                if ((unsigned)ix >= (unsigned)HIN) continue;
                s += ip[iy * HIN + ix] * wp[ky * 4 + kx];
            }
        }
    }
    out[idx] = s;
}

// =====================================================================
// BatchNorm (biased var over (B,H,W)) + ReLU, in place. block per channel
// =====================================================================
__global__ void bn_relu_kernel(float* __restrict__ x, const float* __restrict__ g,
                               const float* __restrict__ be, int C, int B, int HW)
{
    __shared__ float s_sum[256], s_sq[256];
    int c = blockIdx.x;
    int cnt = B * HW;
    float sum = 0.f, sq = 0.f;
    for (int i = threadIdx.x; i < cnt; i += blockDim.x) {
        int b = i / HW, j = i % HW;
        float v = x[(size_t)(b * C + c) * HW + j];
        sum += v; sq += v * v;
    }
    s_sum[threadIdx.x] = sum; s_sq[threadIdx.x] = sq;
    __syncthreads();
    for (int st = 128; st > 0; st >>= 1) {
        if ((int)threadIdx.x < st) {
            s_sum[threadIdx.x] += s_sum[threadIdx.x + st];
            s_sq [threadIdx.x] += s_sq [threadIdx.x + st];
        }
        __syncthreads();
    }
    float mean = s_sum[0] / (float)cnt;
    float var  = s_sq[0] / (float)cnt - mean * mean;
    float istd = rsqrtf(var + 1e-5f);
    float gc = g[c], bc = be[c];
    for (int i = threadIdx.x; i < cnt; i += blockDim.x) {
        int b = i / HW, j = i % HW;
        size_t off = (size_t)(b * C + c) * HW + j;
        float v = gc * (x[off] - mean) * istd + bc;
        x[off] = fmaxf(v, 0.f);
    }
}

// =====================================================================
// Weight packing (f32 -> f16)
// =====================================================================
__global__ void pack_transpose_f16(const float* __restrict__ W, _Float16* __restrict__ Bp,
                                   int N, int K)   // W is [N][K], Bp is [K][N]
{
    int idx = blockIdx.x * blockDim.x + threadIdx.x;
    if (idx >= N * K) return;
    int k = idx / N, n = idx % N;
    Bp[idx] = (_Float16)W[(size_t)n * K + k];
}

__global__ void pack_cast_f16(const float* __restrict__ W, _Float16* __restrict__ Bp, int count)
{
    int idx = blockIdx.x * blockDim.x + threadIdx.x;
    if (idx < count) Bp[idx] = (_Float16)W[idx];
}

// deconv weights per output-parity class. w: [CIN][COUT][4][4] -> Bp: [4][CIN*4][NPAD]
__global__ void pack_deconv_f16(const float* __restrict__ w, _Float16* __restrict__ Bp,
                                int CIN, int COUT, int NPAD)
{
    int K = CIN * 4;
    int total = 4 * K * NPAD;
    int idx = blockIdx.x * blockDim.x + threadIdx.x;
    if (idx >= total) return;
    int p = idx / (K * NPAD); int rem = idx % (K * NPAD);
    int k = rem / NPAD;       int n = rem % NPAD;
    int py = p >> 1, px = p & 1;
    int c = k >> 2, jy = (k >> 1) & 1, jx = k & 1;
    int ky = py ? (jy ? 2 : 0) : (jy ? 3 : 1);
    int kx = px ? (jx ? 2 : 0) : (jx ? 3 : 1);
    _Float16 v = (_Float16)0.f;
    if (n < COUT) v = (_Float16)w[((size_t)(c * COUT + n) * 4 + ky) * 4 + kx];
    Bp[idx] = v;
}

// =====================================================================
// WMMA GEMM: C[M,N] = act( A[M,K](f32) x Bp[K,N](f16) + bias[n/bias_div] )
// requires M%64==0, N%16==0, K%32==0.  block=128 (4 waves), tile 64x16.
// ACT: 0 none, 1 relu, 2 sigmoid
// =====================================================================
template<int ACT>
__global__ void gemm_wmma(const float* __restrict__ A, const _Float16* __restrict__ Bp,
                          const float* __restrict__ bias, float* __restrict__ C,
                          int M, int N, int K, int bias_div)
{
    __shared__ _Float16 As[64][40];
    __shared__ _Float16 BsT[16][40];
    const int lane = threadIdx.x & 31;
    const int wid  = threadIdx.x >> 5;
    const int mBase = blockIdx.x * 64;
    const int nBase = blockIdx.y * 16;
    const int rowL = threadIdx.x >> 1;
    const int col0 = (threadIdx.x & 1) * 16;
    const int gm   = mBase + rowL;

    v8f acc = {};
    for (int k0 = 0; k0 < K; k0 += 32) {
        {
            const v4f* ap = (const v4f*)(A + (size_t)gm * K + k0 + col0);
            v4f f0 = ap[0], f1 = ap[1], f2 = ap[2], f3 = ap[3];
            v8h lo, hi;
            #pragma unroll
            for (int i = 0; i < 4; ++i) {
                lo[i]     = (_Float16)f0[i];
                lo[4 + i] = (_Float16)f1[i];
                hi[i]     = (_Float16)f2[i];
                hi[4 + i] = (_Float16)f3[i];
            }
            *(v8h*)&As[rowL][col0]     = lo;
            *(v8h*)&As[rowL][col0 + 8] = hi;
        }
        stage_b(Bp, N, k0, nBase, BsT, threadIdx.x);
        __syncthreads();
        v16h af = load_a_frag(As, wid, lane);
        v16h bf = load_b_frag(BsT, lane);
        acc = __builtin_amdgcn_wmma_f32_16x16x32_f16(false, af, false, bf,
                                                     (short)0, acc, false, false);
        __syncthreads();
    }

    int n = nBase + (lane & 15);
    float bv = bias[n / bias_div];
    #pragma unroll
    for (int v = 0; v < 8; ++v) {
        int m = mBase + wid * 16 + v + ((lane >> 4) << 3);
        float x = acc[v] + bv;
        if (ACT == 1) x = fmaxf(x, 0.f);
        else if (ACT == 2) x = 1.f / (1.f + __expf(-x));
        C[(size_t)m * N + n] = x;
    }
}

// =====================================================================
// WMMA implicit-GEMM forward conv, k=4, s=2, p=1 (encoder conv2..4).
// K = CIN*16, M = 128*HOUT*HOUT.  Clamp+mask gather (branchless).
// Bp: [K][COUT] = pack_transpose of w[COUT][CIN*16].
// =====================================================================
template<int CIN, int COUT, int HIN, int HOUT>
__global__ void conv_wmma(const float* __restrict__ in, const _Float16* __restrict__ Bp,
                          const float* __restrict__ bias, float* __restrict__ out)
{
    constexpr int K = CIN * 16;
    __shared__ _Float16 As[64][40];
    __shared__ _Float16 BsT[16][40];
    const int lane = threadIdx.x & 31;
    const int wid  = threadIdx.x >> 5;
    const int mBase = blockIdx.x * 64;
    const int nBase = blockIdx.y * 16;
    const int rowL = threadIdx.x >> 1;
    const int col0 = (threadIdx.x & 1) * 16;

    const int gm = mBase + rowL;
    const int b  = gm / (HOUT * HOUT);
    const int rm = gm % (HOUT * HOUT);
    const int oy = rm / HOUT, ox = rm % HOUT;

    // tap validity is K-invariant: precompute clamped coords + 0/1 masks
    float my[4], mx[4];
    int   iyc[4], ixc[4];
    #pragma unroll
    for (int kk = 0; kk < 4; ++kk) {
        int iy = oy * 2 - 1 + kk;
        bool vy = (unsigned)iy < (unsigned)HIN;
        my[kk] = vy ? 1.f : 0.f; iyc[kk] = vy ? iy : 0;
        int ix = ox * 2 - 1 + kk;
        bool vx = (unsigned)ix < (unsigned)HIN;
        mx[kk] = vx ? 1.f : 0.f; ixc[kk] = vx ? ix : 0;
    }
    const float* ib = in + (size_t)b * CIN * HIN * HIN;

    v8f acc = {};
    for (int k0 = 0; k0 < K; k0 += 32) {
        {
            // (k0+col0) is a multiple of 16 -> c fixed, tap index = i
            int c = (k0 + col0) >> 4;
            const float* base = ib + (size_t)c * HIN * HIN;
            v8h lo, hi;
            #pragma unroll
            for (int i = 0; i < 16; ++i) {
                int ky = i >> 2, kx = i & 3;
                float v = my[ky] * mx[kx] * base[iyc[ky] * HIN + ixc[kx]];
                if (i < 8) lo[i] = (_Float16)v; else hi[i - 8] = (_Float16)v;
            }
            *(v8h*)&As[rowL][col0]     = lo;
            *(v8h*)&As[rowL][col0 + 8] = hi;
        }
        stage_b(Bp, COUT, k0, nBase, BsT, threadIdx.x);
        __syncthreads();
        v16h af = load_a_frag(As, wid, lane);
        v16h bf = load_b_frag(BsT, lane);
        acc = __builtin_amdgcn_wmma_f32_16x16x32_f16(false, af, false, bf,
                                                     (short)0, acc, false, false);
        __syncthreads();
    }

    int o = nBase + (lane & 15);
    float bv = bias[o];
    #pragma unroll
    for (int v = 0; v < 8; ++v) {
        int m = mBase + wid * 16 + v + ((lane >> 4) << 3);
        int bi = m / (HOUT * HOUT);
        int rr = m % (HOUT * HOUT);
        int yy = rr / HOUT, xx = rr % HOUT;
        out[((size_t)(bi * COUT + o) * HOUT + yy) * HOUT + xx] = acc[v] + bv;
    }
}

// =====================================================================
// WMMA implicit-GEMM transposed conv (k=4, s=2, p=1), parity-decomposed.
// taps: py==0 -> (ky=1,dy=0),(ky=3,dy=-1); py==1 -> (ky=0,dy=+1),(ky=2,dy=0)
// grid: (MTOT/64, NPAD/16, 4);  NB = 2048
// =====================================================================
template<int CIN, int COUT, int HIN, int ACT>
__global__ void deconv_wmma(const float* __restrict__ in, const _Float16* __restrict__ Bp,
                            const float* __restrict__ bias, float* __restrict__ out)
{
    constexpr int K    = CIN * 4;
    constexpr int NPAD = (COUT < 16) ? 16 : COUT;
    constexpr int HOUT = HIN * 2;

    __shared__ _Float16 As[64][40];
    __shared__ _Float16 BsT[16][40];
    const int lane = threadIdx.x & 31;
    const int wid  = threadIdx.x >> 5;
    const int mBase = blockIdx.x * 64;
    const int nBase = blockIdx.y * 16;
    const int pz = blockIdx.z;
    const int py = pz >> 1, px = pz & 1;
    const int dy0 = py ? 1 : 0, dy1 = py ? 0 : -1;
    const int dx0 = px ? 1 : 0, dx1 = px ? 0 : -1;
    const _Float16* Bpp = Bp + (size_t)pz * K * NPAD;

    const int rowL = threadIdx.x >> 1;
    const int col0 = (threadIdx.x & 1) * 16;
    const int gm = mBase + rowL;
    const int nimg = gm / (HIN * HIN);
    const int rm = gm % (HIN * HIN);
    const int sy = rm / HIN, sx = rm % HIN;

    float myv[2], mxv[2];
    int   iyv[2], ixv[2];
    {
        int y0 = sy + dy0, y1 = sy + dy1;
        bool v0 = (unsigned)y0 < (unsigned)HIN, v1 = (unsigned)y1 < (unsigned)HIN;
        myv[0] = v0 ? 1.f : 0.f; iyv[0] = v0 ? y0 : 0;
        myv[1] = v1 ? 1.f : 0.f; iyv[1] = v1 ? y1 : 0;
        int x0 = sx + dx0, x1 = sx + dx1;
        bool w0 = (unsigned)x0 < (unsigned)HIN, w1 = (unsigned)x1 < (unsigned)HIN;
        mxv[0] = w0 ? 1.f : 0.f; ixv[0] = w0 ? x0 : 0;
        mxv[1] = w1 ? 1.f : 0.f; ixv[1] = w1 ? x1 : 0;
    }
    const float* ib = in + (size_t)nimg * CIN * HIN * HIN;

    v8f acc = {};
    for (int k0 = 0; k0 < K; k0 += 32) {
        {
            int c0 = (k0 + col0) >> 2;   // (k0+col0) multiple of 16
            v8h lo, hi;
            #pragma unroll
            for (int i = 0; i < 16; ++i) {
                int c = c0 + (i >> 2);
                int jy = (i >> 1) & 1, jx = i & 1;
                float v = myv[jy] * mxv[jx] *
                          ib[(size_t)c * HIN * HIN + iyv[jy] * HIN + ixv[jx]];
                if (i < 8) lo[i] = (_Float16)v; else hi[i - 8] = (_Float16)v;
            }
            *(v8h*)&As[rowL][col0]     = lo;
            *(v8h*)&As[rowL][col0 + 8] = hi;
        }
        stage_b(Bpp, NPAD, k0, nBase, BsT, threadIdx.x);
        __syncthreads();
        v16h af = load_a_frag(As, wid, lane);
        v16h bf = load_b_frag(BsT, lane);
        acc = __builtin_amdgcn_wmma_f32_16x16x32_f16(false, af, false, bf,
                                                     (short)0, acc, false, false);
        __syncthreads();
    }

    int nc = nBase + (lane & 15);
    if (nc < COUT) {
        float bv = bias[nc];
        #pragma unroll
        for (int v = 0; v < 8; ++v) {
            int m = mBase + wid * 16 + v + ((lane >> 4) << 3);
            int ni = m / (HIN * HIN);
            int rr = m % (HIN * HIN);
            int syo = rr / HIN, sxo = rr % HIN;
            int oy = 2 * syo + py, ox = 2 * sxo + px;
            float x = acc[v] + bv;
            if (ACT == 1) x = fmaxf(x, 0.f);
            else if (ACT == 2) x = 1.f / (1.f + __expf(-x));
            out[((size_t)(ni * COUT + nc) * HOUT + oy) * HOUT + ox] = x;
        }
    }
}

// =====================================================================
// Small fp32 head kernels
// =====================================================================
__global__ void linear_kernel(const float* __restrict__ A, const float* __restrict__ W,
                              const float* __restrict__ b, float* __restrict__ out,
                              int M, int N, int K)
{
    int idx = blockIdx.x * blockDim.x + threadIdx.x;
    if (idx >= M * N) return;
    int m = idx / N, n = idx % N;
    float s = b[n];
    const float* a = A + (size_t)m * K;
    const float* w = W + (size_t)n * K;
    for (int k = 0; k < K; ++k) s += a[k] * w[k];
    out[idx] = s;
}

__global__ void z_kernel(const float* __restrict__ oz, const float* __restrict__ eps_z,
                         float* __restrict__ z, float* __restrict__ muz, float* __restrict__ varz)
{
    int idx = blockIdx.x * blockDim.x + threadIdx.x;
    if (idx >= 128 * 32) return;
    int b = idx / 32, j = idx % 32;
    float mu = oz[b * 64 + j];
    float v  = __expf(tanhf(oz[b * 64 + 32 + j]));
    muz[idx] = mu;
    varz[idx] = v;
    z[idx] = mu + eps_z[idx] * sqrtf(v);
}

__global__ void w_kernel(const float* __restrict__ ow, const float* __restrict__ eps_w,
                         float* __restrict__ wv, float* __restrict__ muw, float* __restrict__ varw)
{
    int j = threadIdx.x;
    if (j >= 32) return;
    float s1 = 0.f, s2 = 0.f;
    for (int b = 0; b < 128; ++b) {
        float p = __expf(-tanhf(ow[b * 64 + 32 + j]));
        s1 += p;
        s2 += p * ow[b * 64 + j];
    }
    float var = 1.f / s1;
    float mu  = var * s2;
    muw[j] = mu;
    varw[j] = var;
    wv[j] = mu + eps_w[j] * sqrtf(var);
}

__global__ void hb1_kernel(const float* __restrict__ Wb1, const float* __restrict__ bb1,
                           const float* __restrict__ wv, float* __restrict__ h1)
{
    int idx = blockIdx.x * blockDim.x + threadIdx.x;
    if (idx >= 16 * 256) return;
    float s = bb1[idx];
    const float* wp = Wb1 + (size_t)idx * 32;
    for (int i = 0; i < 32; ++i) s += wp[i] * wv[i];
    h1[idx] = fmaxf(s, 0.f);
}

__global__ void ob_kernel(const float* __restrict__ Wb2, const float* __restrict__ bb2,
                          const float* __restrict__ h1, float* __restrict__ ob)
{
    int idx = blockIdx.x * blockDim.x + threadIdx.x;
    if (idx >= 16 * 64) return;
    int k = idx / 64;
    float s = bb2[idx];
    const float* wp = Wb2 + (size_t)idx * 256;
    const float* hp = h1 + (size_t)k * 256;
    for (int o = 0; o < 256; ++o) s += wp[o] * hp[o];
    ob[idx] = s;
}

__global__ void betas_kernel(const float* __restrict__ ob, const float* __restrict__ eps_beta,
                             float* __restrict__ betas, float* __restrict__ musb,
                             float* __restrict__ varsb)
{
    int idx = blockIdx.x * blockDim.x + threadIdx.x;
    if (idx >= 16 * 32) return;
    int k = idx / 32, j = idx % 32;
    float mu = ob[k * 64 + j];
    float v  = __expf(tanhf(ob[k * 64 + 32 + j]));
    musb[idx] = mu;
    varsb[idx] = v;
    betas[idx] = mu + eps_beta[idx] * sqrtf(v);
}

__global__ void hm_kernel(const float* __restrict__ t, float* __restrict__ hm)
{
    int n = threadIdx.x;
    if (n >= 256) return;
    float s = 0.f;
    for (int b = 0; b < 128; ++b) s += t[b * 256 + n];
    hm[n] = s * (1.f / 128.f);
}

__global__ void a_kernel(const float* __restrict__ hm, const float* __restrict__ wv,
                         const float* __restrict__ Wa1, const float* __restrict__ ba1,
                         float* __restrict__ av)
{
    int n = threadIdx.x;
    if (n >= 256) return;
    float s = ba1[n];
    const float* wp = Wa1 + (size_t)n * 288;
    for (int i = 0; i < 256; ++i) s += hm[i] * wp[i];
    for (int i = 0; i < 32; ++i)  s += wv[i] * wp[256 + i];
    av[n] = tanhf(s);
}

__global__ void pi_kernel(const float* __restrict__ av, const float* __restrict__ Wa2,
                          const float* __restrict__ ba2, float* __restrict__ pi)
{
    __shared__ float lg[16];
    int k = threadIdx.x;
    float s = ba2[k];
    for (int n = 0; n < 256; ++n) s += av[n] * Wa2[k * 256 + n];
    lg[k] = s;
    __syncthreads();
    float mx = lg[0];
    for (int i = 1; i < 16; ++i) mx = fmaxf(mx, lg[i]);
    float den = 0.f;
    for (int i = 0; i < 16; ++i) den += __expf(lg[i] - mx);
    pi[k] = __expf(s - mx) / den;
}

__global__ void decin_kernel(const float* __restrict__ z, const float* __restrict__ betas,
                             float* __restrict__ decin)
{
    int idx = blockIdx.x * blockDim.x + threadIdx.x;
    if (idx >= 2048 * 64) return;
    int r = idx / 64, c = idx % 64;
    int k = r / 128, b = r % 128;
    decin[idx] = (c < 32) ? z[b * 32 + c] : betas[k * 32 + (c - 32)];
}

// =====================================================================
// Host side
// =====================================================================
extern "C" void kernel_launch(void* const* d_in, const int* in_sizes, int n_in,
                              void* d_out, int out_size, void* d_ws, size_t ws_size,
                              hipStream_t stream)
{
    (void)in_sizes; (void)n_in; (void)out_size; (void)ws_size;
    const float* x    = (const float*)d_in[0];
    const float* c1w  = (const float*)d_in[1];  const float* c1b = (const float*)d_in[2];
    const float* g1   = (const float*)d_in[3];  const float* be1 = (const float*)d_in[4];
    const float* c2w  = (const float*)d_in[5];  const float* c2b = (const float*)d_in[6];
    const float* g2   = (const float*)d_in[7];  const float* be2 = (const float*)d_in[8];
    const float* c3w  = (const float*)d_in[9];  const float* c3b = (const float*)d_in[10];
    const float* g3   = (const float*)d_in[11]; const float* be3 = (const float*)d_in[12];
    const float* c4w  = (const float*)d_in[13]; const float* c4b = (const float*)d_in[14];
    const float* g4   = (const float*)d_in[15]; const float* be4 = (const float*)d_in[16];
    const float* c5w  = (const float*)d_in[17]; const float* c5b = (const float*)d_in[18];
    const float* g5   = (const float*)d_in[19]; const float* be5 = (const float*)d_in[20];
    const float* Wz   = (const float*)d_in[21]; const float* bz  = (const float*)d_in[22];
    const float* Ww   = (const float*)d_in[23]; const float* bw  = (const float*)d_in[24];
    const float* Wax  = (const float*)d_in[25]; const float* bax = (const float*)d_in[26];
    const float* Wa1  = (const float*)d_in[27]; const float* ba1 = (const float*)d_in[28];
    const float* Wa2  = (const float*)d_in[29]; const float* ba2 = (const float*)d_in[30];
    const float* Wb1  = (const float*)d_in[31]; const float* bb1 = (const float*)d_in[32];
    const float* Wb2  = (const float*)d_in[33]; const float* bb2 = (const float*)d_in[34];
    const float* Wd   = (const float*)d_in[35]; const float* bd  = (const float*)d_in[36];
    const float* d1w  = (const float*)d_in[37]; const float* d1b = (const float*)d_in[38];
    const float* d2w  = (const float*)d_in[39]; const float* d2b = (const float*)d_in[40];
    const float* d3w  = (const float*)d_in[41]; const float* d3b = (const float*)d_in[42];
    const float* d4w  = (const float*)d_in[43]; const float* d4b = (const float*)d_in[44];
    const float* d5w  = (const float*)d_in[45]; const float* d5b = (const float*)d_in[46];
    const float* eps_z    = (const float*)d_in[47];
    const float* eps_w    = (const float*)d_in[48];
    const float* eps_beta = (const float*)d_in[49];

    // -------- output layout (floats) --------
    float* out   = (float*)d_out;
    float* musx  = out;                     // 25165824
    float* muz   = out + 25165824;          // 4096
    float* varz  = muz + 4096;              // 4096
    float* musb  = varz + 4096;             // 512
    float* varsb = musb + 512;              // 512
    float* muw   = varsb + 512;             // 32
    float* varw  = muw + 32;                // 32
    float* piO   = varw + 32;               // 16

    // -------- workspace layout (aliased regions) --------
    char* wsb = (char*)d_ws;
    float* rBig = (float*)wsb;                                   // y1 / a2 / a4
    float* rMid = (float*)(wsb + (size_t)268435456);             // y2 / a1 / a3
    char*  cur  = wsb + (size_t)268435456 + (size_t)67108864;
    auto takef = [&](size_t n) -> float* {
        float* p = (float*)cur; cur += ((n * 4 + 255) & ~(size_t)255); return p;
    };
    auto takeh = [&](size_t n) -> _Float16* {
        _Float16* p = (_Float16*)cur; cur += ((n * 2 + 255) & ~(size_t)255); return p;
    };
    float* y1 = rBig;  float* a2v = rBig;  float* a4v = rBig;
    float* y2 = rMid;  float* a1v = rMid;  float* a3v = rMid;
    float* y3    = takef(524288);
    float* y4    = takef(131072);
    float* h     = takef(32768);
    float* oz    = takef(8192);
    float* ow    = takef(8192);
    float* tb    = takef(32768);
    float* hm    = takef(256);
    float* wv    = takef(32);
    float* h1    = takef(4096);
    float* obv   = takef(1024);
    float* av    = takef(256);
    float* zb    = takef(4096);
    float* betas = takef(512);
    float* decin = takef(131072);
    float* dfc   = takef(524288);
    _Float16* pc2 = takeh(512 * 32);
    _Float16* pc3 = takeh(512 * 64);
    _Float16* pc4 = takeh(1024 * 64);
    _Float16* p5  = takeh(1024 * 256);
    _Float16* pWd = takeh(64 * 256);
    _Float16* pd1 = takeh(256 * 1024);
    _Float16* pd2 = takeh(4 * 256 * 64);
    _Float16* pd3 = takeh(4 * 256 * 32);
    _Float16* pd4 = takeh(4 * 128 * 32);
    _Float16* pd5 = takeh(4 * 128 * 16);

    const int TB = 256;
    auto cdiv = [](int a, int b) { return (a + b - 1) / b; };

    // -------------------- encoder --------------------
    conv_s2_kernel<<<cdiv(128*32*32*32, TB), TB, 0, stream>>>(x, c1w, c1b, y1, 128, 3, 32, 64, 32);
    bn_relu_kernel<<<32,  TB, 0, stream>>>(y1, g1, be1, 32, 128, 32*32);

    pack_transpose_f16<<<cdiv(32*512, TB), TB, 0, stream>>>(c2w, pc2, 32, 512);
    { dim3 g(32768/64, 32/16); conv_wmma<32, 32, 32, 16><<<g, 128, 0, stream>>>(y1, pc2, c2b, y2); }
    bn_relu_kernel<<<32,  TB, 0, stream>>>(y2, g2, be2, 32, 128, 16*16);

    pack_transpose_f16<<<cdiv(64*512, TB), TB, 0, stream>>>(c3w, pc3, 64, 512);
    { dim3 g(8192/64, 64/16); conv_wmma<32, 64, 16, 8><<<g, 128, 0, stream>>>(y2, pc3, c3b, y3); }
    bn_relu_kernel<<<64,  TB, 0, stream>>>(y3, g3, be3, 64, 128, 8*8);

    pack_transpose_f16<<<cdiv(64*1024, TB), TB, 0, stream>>>(c4w, pc4, 64, 1024);
    { dim3 g(2048/64, 64/16); conv_wmma<64, 64, 8, 4><<<g, 128, 0, stream>>>(y3, pc4, c4b, y4); }
    bn_relu_kernel<<<64,  TB, 0, stream>>>(y4, g4, be4, 64, 128, 4*4);

    // conv5 as WMMA GEMM: M=128, K=1024, N=256
    pack_transpose_f16<<<cdiv(256*1024, TB), TB, 0, stream>>>(c5w, p5, 256, 1024);
    { dim3 g(128/64, 256/16); gemm_wmma<0><<<g, 128, 0, stream>>>(y4, p5, c5b, h, 128, 256, 1024, 1); }
    bn_relu_kernel<<<256, TB, 0, stream>>>(h, g5, be5, 256, 128, 1);

    // -------------------- latent heads --------------------
    linear_kernel<<<cdiv(128*64, TB), TB, 0, stream>>>(h, Wz, bz, oz, 128, 64, 256);
    z_kernel<<<cdiv(4096, TB), TB, 0, stream>>>(oz, eps_z, zb, muz, varz);
    linear_kernel<<<cdiv(128*64, TB), TB, 0, stream>>>(h, Ww, bw, ow, 128, 64, 256);
    w_kernel<<<1, 32, 0, stream>>>(ow, eps_w, wv, muw, varw);
    hb1_kernel<<<cdiv(16*256, TB), TB, 0, stream>>>(Wb1, bb1, wv, h1);
    ob_kernel<<<cdiv(16*64, TB), TB, 0, stream>>>(Wb2, bb2, h1, obv);
    betas_kernel<<<cdiv(512, TB), TB, 0, stream>>>(obv, eps_beta, betas, musb, varsb);
    linear_kernel<<<cdiv(128*256, TB), TB, 0, stream>>>(h, Wax, bax, tb, 128, 256, 256);
    hm_kernel<<<1, 256, 0, stream>>>(tb, hm);
    a_kernel<<<1, 256, 0, stream>>>(hm, wv, Wa1, ba1, av);
    pi_kernel<<<1, 16, 0, stream>>>(av, Wa2, ba2, piO);

    // -------------------- decoder --------------------
    decin_kernel<<<cdiv(2048*64, TB), TB, 0, stream>>>(zb, betas, decin);

    pack_transpose_f16<<<cdiv(256*64, TB), TB, 0, stream>>>(Wd, pWd, 256, 64);
    { dim3 g(2048/64, 256/16); gemm_wmma<1><<<g, 128, 0, stream>>>(decin, pWd, bd, dfc, 2048, 256, 64, 1); }

    pack_cast_f16<<<cdiv(256*1024, TB), TB, 0, stream>>>(d1w, pd1, 256 * 1024);
    { dim3 g(2048/64, 1024/16); gemm_wmma<1><<<g, 128, 0, stream>>>(dfc, pd1, d1b, a1v, 2048, 1024, 256, 16); }

    pack_deconv_f16<<<cdiv(4*256*64, TB), TB, 0, stream>>>(d2w, pd2, 64, 64, 64);
    { dim3 g((2048*4*4)/64, 64/16, 4); deconv_wmma<64, 64, 4, 1><<<g, 128, 0, stream>>>(a1v, pd2, d2b, a2v); }

    pack_deconv_f16<<<cdiv(4*256*32, TB), TB, 0, stream>>>(d3w, pd3, 64, 32, 32);
    { dim3 g((2048*8*8)/64, 32/16, 4); deconv_wmma<64, 32, 8, 1><<<g, 128, 0, stream>>>(a2v, pd3, d3b, a3v); }

    pack_deconv_f16<<<cdiv(4*128*32, TB), TB, 0, stream>>>(d4w, pd4, 32, 32, 32);
    { dim3 g((2048*16*16)/64, 32/16, 4); deconv_wmma<32, 32, 16, 1><<<g, 128, 0, stream>>>(a3v, pd4, d4b, a4v); }

    pack_deconv_f16<<<cdiv(4*128*16, TB), TB, 0, stream>>>(d5w, pd5, 32, 3, 16);
    { dim3 g((2048*32*32)/64, 16/16, 4); deconv_wmma<32, 3, 32, 2><<<g, 128, 0, stream>>>(a4v, pd5, d5b, musx); }
}
